// LSTMModel_40853728920216
// MI455X (gfx1250) — matrix operand order repaired and checked by reading, not compiled, not run on hardware
//
#include <hip/hip_runtime.h>

typedef __attribute__((ext_vector_type(16))) _Float16 v16h;
typedef __attribute__((ext_vector_type(8)))  float    v8f;

#define T_LEN 2048
#define B_TOT 2048
#define H_DIM 25

__device__ __forceinline__ float fast_sigmoid(float x) {
    // 1 / (1 + 2^(-x*log2e))
    float t = __builtin_amdgcn_exp2f(-1.44269504088896340736f * x);
    return __builtin_amdgcn_rcpf(1.0f + t);
}
__device__ __forceinline__ float fast_tanh(float x) {
    // 2/(1 + 2^(-2x*log2e)) - 1
    float t = __builtin_amdgcn_exp2f(-2.88539008177792681472f * x);
    return __builtin_fmaf(2.0f, __builtin_amdgcn_rcpf(1.0f + t), -1.0f);
}

// K-slot -> hidden-unit permutation: keeps the B-matrix build lane-local
// (D-tile lane-half ownership of unit j is (j>>3)&1; this code's B puts K 0..15 in
//  lanes 0-15 and K 16..31 in lanes 16-31 — this bijection reconciles them).
__device__ __forceinline__ int kslot_to_unit(int k) {
    return (k < 8) ? k : (k < 16) ? (k + 8) : (k < 24) ? (k - 8) : 24;
}

__global__ __launch_bounds__(32)
void lstm_fused(const float* __restrict__ x,
                const float* __restrict__ w_ih,
                const float* __restrict__ w_hh,
                const float* __restrict__ b_ih,
                const float* __restrict__ b_hh,
                const float* __restrict__ w_dense,
                const float* __restrict__ b_dense,
                float* __restrict__ out)
{
    const int lane = threadIdx.x & 31;
    const int half = lane >> 4;          // lane half (A/B K-group, D M-offset)
    const int ncol = lane & 15;          // batch column within the 16-wide tile
    const int bcol = blockIdx.x * 16 + ncol;

    // ---------------- A tiles (f16), built once ----------------------------
    // Padded rows: row = 32*gate + j (j<25 valid). K columns: 0..24 = W_hh
    // (unit-permuted), 25 = combined bias, 26 = w_ih (IN==1 input proj).
    // A 16x32 f16 layout: M = lane&15 ; VGPR v packs K = 16*half+2*v+p.
    v16h A[8];
    {
        const int M = ncol;
        #pragma unroll
        for (int tm = 0; tm < 8; ++tm) {
            const int row = tm * 16 + M;
            const int g   = row >> 5;
            const int j   = row & 31;
            const bool rv = (j < H_DIM);
            const int grow = g * H_DIM + j;        // row of w_hh [100,25]
            float bias = 0.f, wih = 0.f;
            if (rv) { bias = b_ih[grow] + b_hh[grow]; wih = w_ih[grow]; }
            v16h a;
            #pragma unroll
            for (int v = 0; v < 8; ++v) {
                #pragma unroll
                for (int p = 0; p < 2; ++p) {
                    const int K = 16 * half + 2 * v + p;
                    float val = 0.f;
                    if (rv) {
                        if (K < 25)       val = w_hh[grow * H_DIM + kslot_to_unit(K)];
                        else if (K == 25) val = bias;
                        else if (K == 26) val = wih;
                    }
                    a[2 * v + p] = (_Float16)val;
                }
            }
            A[tm] = a;
        }
    }

    // Dense-head weights for the units this lane owns (slot j = 16*tHi+8*half+e)
    float wdl[16];
    #pragma unroll
    for (int tHi = 0; tHi < 2; ++tHi)
        #pragma unroll
        for (int e = 0; e < 8; ++e) {
            const int j = 16 * tHi + 8 * half + e;
            wdl[8 * tHi + e] = (j < H_DIM) ? w_dense[j] : 0.f;
        }
    const float bd = b_dense[0];

    float hloc[16], cloc[16];
    #pragma unroll
    for (int i = 0; i < 16; ++i) { hloc[i] = 0.f; cloc[i] = 0.f; }

    const float* __restrict__ xrow = x   + (size_t)bcol * T_LEN;   // x[b, t, 0]
    float*       __restrict__ orow = out + (size_t)bcol * T_LEN;   // out[b, t, 0]

    #pragma unroll 1
    for (int t = 0; t < T_LEN; ++t) {
        const float xv = xrow[t];
        __builtin_prefetch(xrow + t + 64, 0, 0);   // global_prefetch_b8

        // ---- B (f16) 32x16: lanes 0-15 supply K=0..15, lanes 16-31 K=16..31.
        // With the unit permutation, local slot kp maps straight to hloc[kp];
        // only the high half carries the constant-1 (bias) and x(t) K-slots.
        v16h Bm;
        #pragma unroll
        for (int v = 0; v < 8; ++v) {
            #pragma unroll
            for (int p = 0; p < 2; ++p) {
                const int kp = 2 * v + p;          // local K within this half
                float val = hloc[kp];
                if (kp == 9)  val = half ? 1.0f : val;   // K=25 -> bias * 1
                if (kp == 10) val = half ? xv   : val;   // K=26 -> w_ih * x(t)
                if (kp >= 11) val = half ? 0.0f : val;   // K=27..31 padding
                Bm[kp] = (_Float16)val;
            }
        }

        // ---- gates^T = A * B  (8 independent 16x16x32 WMMAs, C = 0) ----
        v8f dz = {};
        v8f d[8];
        #pragma unroll
        for (int tm = 0; tm < 8; ++tm) {
            d[tm] = __builtin_amdgcn_wmma_f32_16x16x32_f16(
                false, A[tm], false, Bm, (short)0, dz, false, false);
        }

        // ---- per-lane LSTM cell update + dense-output partial ----
        // Unit j = 16*tHi + 8*half + e: i/f/g/o live at tiles tHi, 2+tHi,
        // 4+tHi, 6+tHi, element e — same lane, pure register math.
        float partial = 0.f;
        #pragma unroll
        for (int tHi = 0; tHi < 2; ++tHi) {
            #pragma unroll
            for (int e = 0; e < 8; ++e) {
                const int idx = 8 * tHi + e;
                const float ip = d[0 + tHi][e];
                const float fp = d[2 + tHi][e];
                const float gp = d[4 + tHi][e];
                const float op = d[6 + tHi][e];
                const float iv = fast_sigmoid(ip);
                const float fv = fast_sigmoid(fp);
                const float gv = fast_tanh(gp);
                const float ov = fast_sigmoid(op);
                const float c  = __builtin_fmaf(fv, cloc[idx], iv * gv);
                const float h  = ov * fast_tanh(c);
                cloc[idx] = c;
                hloc[idx] = h;
                partial = __builtin_fmaf(wdl[idx], h, partial);
            }
        }

        // dense head: sum the two lane halves, add bias, store out[b,t]
        const float other = __shfl_xor(partial, 16, 32);
        if (lane < 16) orow[t] = partial + other + bd;
    }
}

extern "C" void kernel_launch(void* const* d_in, const int* in_sizes, int n_in,
                              void* d_out, int out_size, void* d_ws, size_t ws_size,
                              hipStream_t stream) {
    (void)in_sizes; (void)n_in; (void)out_size; (void)d_ws; (void)ws_size;
    const float* x       = (const float*)d_in[0];
    const float* w_ih    = (const float*)d_in[1];
    const float* w_hh    = (const float*)d_in[2];
    const float* b_ih    = (const float*)d_in[3];
    const float* b_hh    = (const float*)d_in[4];
    const float* w_dense = (const float*)d_in[5];
    const float* b_dense = (const float*)d_in[6];
    float* out = (float*)d_out;

    dim3 grid(B_TOT / 16);   // 128 waves, one 16-batch recurrence tile each
    dim3 block(32);          // one wave32 per block, no barriers needed
    hipLaunchKernelGGL(lstm_fused, grid, block, 0, stream,
                       x, w_ih, w_hh, b_ih, b_hh, w_dense, b_dense, out);
}
